// HyenaBlock_57870389346355
// MI455X (gfx1250) — matrix operand order repaired and checked by reading, compiled, not run on hardware
//
#include <hip/hip_runtime.h>
#include <hip/hip_bf16.h>

#define DMODEL 512
#define HDIM   2048
#define TLEN   16384
#define NBATCH 4
#define KCONV  17

typedef __attribute__((ext_vector_type(16))) __bf16 v16bf;
typedef __attribute__((ext_vector_type(8)))  float  v8f;

union FragU { v16bf v; unsigned int u[8]; };

__device__ __forceinline__ unsigned short f2bf(float x) {
  unsigned int u = __builtin_bit_cast(unsigned int, x);
  u = (u + 0x7FFFu + ((u >> 16) & 1u)) >> 16;
  return (unsigned short)u;
}
__device__ __forceinline__ float bf2f(unsigned short b) {
  unsigned int u = ((unsigned int)b) << 16;
  return __builtin_bit_cast(float, u);
}
__device__ __forceinline__ float gelu_tanh(float x) {
  float x3 = x * x * x;
  return 0.5f * x * (1.0f + tanhf(0.7978845608028654f * (x + 0.044715f * x3)));
}

// One 16-byte async global->LDS copy per lane (ASYNCcnt-tracked).
__device__ __forceinline__ void async_copy16(unsigned lds_off, const void* gptr) {
  asm volatile("global_load_async_to_lds_b128 %0, %1, off"
               :: "v"(lds_off), "v"(gptr) : "memory");
}

// A fragment: 16x32 bf16 tile from row-major buffer (row stride in elements).
// ISA 7.12.2: lanes 0-15 -> M rows, VGPR j holds K pair 2*(j&3) + 8*half + 16*(j>>2)
__device__ __forceinline__ v16bf load_a_frag(const unsigned short* buf, int row0,
                                             int k0, int stride, int lane) {
  const int r = lane & 15, half = lane >> 4;
  FragU f;
#pragma unroll
  for (int j = 0; j < 8; ++j) {
    int kb = ((j & 3) << 1) + (half << 3) + ((j >> 2) << 4);
    f.u[j] = *(const unsigned int*)&buf[(size_t)(row0 + r) * stride + k0 + kb];
  }
  return f.v;
}

// B fragment: 32x16 bf16 (KxN); source packed transposed: buf[n*stride + k].
// VGPR j holds K pair 2*(j&3) + 8*half + 16*(j>>2) (same order as A).
__device__ __forceinline__ v16bf load_b_frag(const unsigned short* buf, int n0,
                                             int k0, int stride, int lane) {
  const int n = lane & 15, half = lane >> 4;
  const unsigned short* p = &buf[(size_t)(n0 + n) * stride + k0 + (half << 3)];
  FragU f;
#pragma unroll
  for (int j = 0; j < 8; ++j) f.u[j] = *(const unsigned int*)&p[((j & 3) << 1) + ((j >> 2) << 4)];
  return f.v;
}

__device__ __forceinline__ v8f wmma_bf16(v16bf a, v16bf b, v8f c) {
  return __builtin_amdgcn_wmma_f32_16x16x32_bf16(false, a, false, b, (short)0, c,
                                                 false, false);
}

// ---------------- weight pack: dst[n*K+k] = bf16(src[k*N+n]) ----------------
__global__ void pack_wT(const float* __restrict__ src,
                        unsigned short* __restrict__ dst, int K, int N) {
  int idx = blockIdx.x * blockDim.x + threadIdx.x;
  if (idx < K * N) {
    int n = idx / K, k = idx - n * K;
    dst[idx] = f2bf(src[(size_t)k * N + n]);
  }
}

// ---------------- kernel 1: rmsnorm -> gate GEMM -> conv -> proj GEMM -------
__global__ __launch_bounds__(256, 1) void hyena_block1(
    const float* __restrict__ x, const float* __restrict__ ln1,
    const float* __restrict__ convw, const unsigned short* __restrict__ gateT,
    const unsigned short* __restrict__ projT, float* __restrict__ out) {
  __shared__ unsigned short hnb[80 * DMODEL];  // 16-row causal halo + 64 rows
  __shared__ unsigned short gb[64 * DMODEL];   // g, then g*conv(hn)

  const int tid = threadIdx.x, lane = tid & 31, wave = tid >> 5;
  const int b = blockIdx.x >> 8;
  const int t0 = (blockIdx.x & 255) * 64;
  const size_t base = ((size_t)b * TLEN + t0) * DMODEL;

  // phase 1: hn = rmsnorm(x) -> bf16 LDS (one wave per row, lanes over channels)
  for (int r = wave; r < 80; r += 8) {
    const int rt = t0 + r - 16;
    if (rt >= 0) {
      const float* xr = x + ((size_t)b * TLEN + rt) * DMODEL;
      float v[16];
      float ss = 0.f;
#pragma unroll
      for (int i = 0; i < 16; ++i) { float t = xr[i * 32 + lane]; v[i] = t; ss += t * t; }
#pragma unroll
      for (int off = 16; off >= 1; off >>= 1) ss += __shfl_xor(ss, off, 32);
      const float inv = rsqrtf(ss * (1.0f / DMODEL) + 1e-6f);
#pragma unroll
      for (int i = 0; i < 16; ++i)
        hnb[r * DMODEL + i * 32 + lane] = f2bf(v[i] * inv * ln1[i * 32 + lane]);
    } else {
#pragma unroll
      for (int i = 0; i < 16; ++i) hnb[r * DMODEL + i * 32 + lane] = 0;
    }
  }
  __syncthreads();

  const int rb = wave >> 1, nh = wave & 1;     // row-block 0..3, N-half 0..1
  const int rlo = (lane >> 4) * 8, c15 = lane & 15;

  // phase 2: g = sigmoid(hn @ gate_w)
  {
    v16bf A[16];
#pragma unroll
    for (int ks = 0; ks < 16; ++ks)
      A[ks] = load_a_frag(hnb, 16 + rb * 16, ks * 32, DMODEL, lane);
    for (int nt = nh * 16; nt < nh * 16 + 16; ++nt) {
      __builtin_prefetch(&gateT[(size_t)(nt * 16 + 16) * DMODEL], 0, 1);
      v8f acc = {};
#pragma unroll
      for (int ks = 0; ks < 16; ++ks) {
        v16bf Bf = load_b_frag(gateT, nt * 16, ks * 32, DMODEL, lane);
        acc = wmma_bf16(A[ks], Bf, acc);
      }
#pragma unroll
      for (int vv = 0; vv < 8; ++vv) {
        float gv = 1.0f / (1.0f + __expf(-acc[vv]));
        gb[(rb * 16 + rlo + vv) * DMODEL + nt * 16 + c15] = f2bf(gv);
      }
    }
  }
  __syncthreads();

  // phase 3: m = g * causal_dwconv(hn), in place in gb (2 channels/thread)
  {
    const int c = tid * 2;
    float w0[KCONV], w1[KCONV];
#pragma unroll
    for (int k = 0; k < KCONV; ++k) {
      w0[k] = convw[c * KCONV + k];
      w1[k] = convw[(c + 1) * KCONV + k];
    }
    for (int r = 0; r < 64; ++r) {
      float s0 = 0.f, s1 = 0.f;
#pragma unroll
      for (int k = 0; k < KCONV; ++k) {
        unsigned int hv = *(const unsigned int*)&hnb[(r + k) * DMODEL + c];
        s0 += w0[k] * bf2f((unsigned short)(hv & 0xFFFFu));
        s1 += w1[k] * bf2f((unsigned short)(hv >> 16));
      }
      unsigned int gv = *(const unsigned int*)&gb[r * DMODEL + c];
      unsigned int m0 = f2bf(s0 * bf2f((unsigned short)(gv & 0xFFFFu)));
      unsigned int m1 = f2bf(s1 * bf2f((unsigned short)(gv >> 16)));
      *(unsigned int*)&gb[r * DMODEL + c] = m0 | (m1 << 16);
    }
  }
  __syncthreads();

  // phase 4: out = x + m @ proj_w
  {
    v16bf A[16];
#pragma unroll
    for (int ks = 0; ks < 16; ++ks)
      A[ks] = load_a_frag(gb, rb * 16, ks * 32, DMODEL, lane);
    for (int nt = nh * 16; nt < nh * 16 + 16; ++nt) {
      __builtin_prefetch(&projT[(size_t)(nt * 16 + 16) * DMODEL], 0, 1);
      v8f acc = {};
#pragma unroll
      for (int ks = 0; ks < 16; ++ks) {
        v16bf Bf = load_b_frag(projT, nt * 16, ks * 32, DMODEL, lane);
        acc = wmma_bf16(A[ks], Bf, acc);
      }
#pragma unroll
      for (int vv = 0; vv < 8; ++vv) {
        size_t idx = base + (size_t)(rb * 16 + rlo + vv) * DMODEL + nt * 16 + c15;
        out[idx] = x[idx] + acc[vv];
      }
    }
  }
}

// ---------------- kernel 2: rmsnorm -> MLP with async-LDS weight staging ----
__global__ __launch_bounds__(256, 1) void hyena_block2(
    float* __restrict__ xio, const float* __restrict__ ln2,
    const unsigned short* __restrict__ w1T, const unsigned short* __restrict__ w2T) {
  __shared__ unsigned short smem[128 * DMODEL];          // 128 KB, phase-aliased
  unsigned short* hnb   = smem;                          // [128][512] phase 1 only
  unsigned short* w2buf = smem;                          // [512][32]  32 KB
  unsigned short* midb  = smem + 512 * 32;               // [128][32]   8 KB
  unsigned short* w1b0  = smem + 512 * 32 + 128 * 32;    // [32][512]  32 KB
  unsigned short* w1b1  = w1b0 + 32 * 512;               // [32][512]  32 KB

  const int tid = threadIdx.x, lane = tid & 31, wave = tid >> 5;
  const int b = blockIdx.x >> 7;
  const int t0 = (blockIdx.x & 127) * 128;
  const size_t base = ((size_t)b * TLEN + t0) * DMODEL;

  // phase 1: hn2 = rmsnorm(x1) -> bf16 LDS
  for (int r = wave; r < 128; r += 8) {
    const float* xr = xio + base + (size_t)r * DMODEL;
    float v[16];
    float ss = 0.f;
#pragma unroll
    for (int i = 0; i < 16; ++i) { float t = xr[i * 32 + lane]; v[i] = t; ss += t * t; }
#pragma unroll
    for (int off = 16; off >= 1; off >>= 1) ss += __shfl_xor(ss, off, 32);
    const float inv = rsqrtf(ss * (1.0f / DMODEL) + 1e-6f);
#pragma unroll
    for (int i = 0; i < 16; ++i)
      hnb[r * DMODEL + i * 32 + lane] = f2bf(v[i] * inv * ln2[i * 32 + lane]);
  }
  __syncthreads();

  const int rlo = (lane >> 4) * 8, c15 = lane & 15;
  const int row0 = wave * 16;

  // preload this wave's 16 A-fragments (invariant across all 64 chunks)
  v16bf A1[16];
#pragma unroll
  for (int ks = 0; ks < 16; ++ks)
    A1[ks] = load_a_frag(hnb, row0, ks * 32, DMODEL, lane);
  __syncthreads();                      // hnb dead -> smem repurposed

  // prologue: async-stage w1 chunk 0 (32 rows x 512 K, 16 B per lane per op)
#pragma unroll
  for (int i = 0; i < 8; ++i) {
    int c = i * 256 + tid;              // 16-byte chunk id 0..2047
    async_copy16((unsigned)(size_t)&w1b0[c * 8],
                 &w1T[(size_t)(c >> 6) * DMODEL + (c & 63) * 8]);
  }

  v8f zero = {};
  v8f acc[32];
#pragma unroll
  for (int nt = 0; nt < 32; ++nt) acc[nt] = zero;

  for (int ch = 0; ch < 64; ++ch) {     // 64 chunks of 32 hidden columns
    const unsigned short* w1cur = (ch & 1) ? w1b1 : w1b0;
    unsigned short*       w1nxt = (ch & 1) ? w1b0 : w1b1;

    asm volatile("s_wait_asynccnt 0x0" ::: "memory");  // w1[ch] landed (this wave)
    __syncthreads();                    // all waves: w1[ch] ready; w2buf/midb free

    // stage w2 rows for chunk ch: w2buf[n*32 + kk] = w2T[n*HDIM + ch*32 + kk]
#pragma unroll
    for (int i = 0; i < 8; ++i) {
      int c = i * 256 + tid;
      async_copy16((unsigned)(size_t)&w2buf[c * 8],
                   &w2T[(size_t)(c >> 2) * HDIM + ch * 32 + (c & 3) * 8]);
    }
    // prefetch w1 chunk ch+1 into the alternate buffer
    if (ch < 63) {
#pragma unroll
      for (int i = 0; i < 8; ++i) {
        int c = i * 256 + tid;
        async_copy16((unsigned)(size_t)&w1nxt[c * 8],
                     &w1T[(size_t)((ch + 1) * 32 + (c >> 6)) * DMODEL + (c & 63) * 8]);
      }
    }

    // GEMM1 (+GELU) from LDS-staged w1
    v8f m0 = zero, m1 = zero;
#pragma unroll
    for (int ks = 0; ks < 16; ++ks) {
      v16bf B0 = load_b_frag(w1cur, 0, ks * 32, DMODEL, lane);
      v16bf B1 = load_b_frag(w1cur, 16, ks * 32, DMODEL, lane);
      m0 = wmma_bf16(A1[ks], B0, m0);
      m1 = wmma_bf16(A1[ks], B1, m1);
    }
#pragma unroll
    for (int vv = 0; vv < 8; ++vv) {
      int rr = row0 + rlo + vv;
      midb[rr * 32 + c15]      = f2bf(gelu_tanh(m0[vv]));
      midb[rr * 32 + 16 + c15] = f2bf(gelu_tanh(m1[vv]));
    }
    // async loads retire in order: <=8 outstanding means the w2 group is done
    if (ch < 63) asm volatile("s_wait_asynccnt 0x8" ::: "memory");
    else         asm volatile("s_wait_asynccnt 0x0" ::: "memory");
    __syncthreads();                    // midb written by all; w2buf staged

    // GEMM2: accumulate out += mid_chunk @ w2[chunk rows, :] (all from LDS)
    v16bf A2 = load_a_frag(midb, row0, 0, 32, lane);
#pragma unroll
    for (int nt = 0; nt < 32; ++nt) {
      v16bf Bf = load_b_frag(w2buf, nt * 16, 0, 32, lane);
      acc[nt] = wmma_bf16(A2, Bf, acc[nt]);
    }
    // top-of-loop wait+barrier protects w2buf/midb reuse next iteration
  }

  // epilogue: out = x1 + acc (in place)
#pragma unroll
  for (int nt = 0; nt < 32; ++nt) {
#pragma unroll
    for (int vv = 0; vv < 8; ++vv) {
      size_t idx = base + (size_t)(row0 + rlo + vv) * DMODEL + nt * 16 + c15;
      xio[idx] += acc[nt][vv];
    }
  }
}

extern "C" void kernel_launch(void* const* d_in, const int* in_sizes, int n_in,
                              void* d_out, int out_size, void* d_ws, size_t ws_size,
                              hipStream_t stream) {
  (void)in_sizes; (void)n_in; (void)out_size; (void)ws_size;
  const float* x     = (const float*)d_in[0];
  const float* ln1   = (const float*)d_in[1];
  const float* gate  = (const float*)d_in[2];
  const float* convw = (const float*)d_in[3];
  const float* proj  = (const float*)d_in[4];
  const float* ln2   = (const float*)d_in[5];
  const float* w1    = (const float*)d_in[6];
  const float* w2    = (const float*)d_in[7];
  float* out = (float*)d_out;

  unsigned short* ws = (unsigned short*)d_ws;
  unsigned short* gateT = ws;                                   // 512*512
  unsigned short* projT = ws + 512 * 512;                       // 512*512
  unsigned short* w1T   = ws + 2 * 512 * 512;                   // 2048*512 (n-major)
  unsigned short* w2T   = ws + 2 * 512 * 512 + 512 * 2048;      // 512*2048 (n-major)

  pack_wT<<<(512 * 512 + 255) / 256, 256, 0, stream>>>(gate, gateT, 512, 512);
  pack_wT<<<(512 * 512 + 255) / 256, 256, 0, stream>>>(proj, projT, 512, 512);
  pack_wT<<<(512 * 2048 + 255) / 256, 256, 0, stream>>>(w1, w1T, 512, 2048);
  pack_wT<<<(2048 * 512 + 255) / 256, 256, 0, stream>>>(w2, w2T, 2048, 512);

  hyena_block1<<<NBATCH * (TLEN / 64), 256, 0, stream>>>(x, ln1, convw, gateT, projT, out);
  hyena_block2<<<NBATCH * (TLEN / 128), 256, 0, stream>>>(out, ln2, w1T, w2T);
}
